// MultiheadAttention_1632087573145
// MI455X (gfx1250) — matrix operand order repaired and checked by reading, not compiled, not run on hardware
//
#include <hip/hip_runtime.h>
#include <hip/hip_bf16.h>

typedef __attribute__((ext_vector_type(16))) __bf16 v16bf;
typedef __attribute__((ext_vector_type(8)))  __bf16 v8bf;
typedef __attribute__((ext_vector_type(8)))  float  v8f;

constexpr int D_  = 1024;
constexpr int T_  = 512;
constexpr int B_  = 32;
constexpr int H_  = 16;
constexpr int DH_ = 64;

// ---------------------------------------------------------------------------
// WMMA helper. Fragment convention used here (contiguous k order), bf16 16x16x32:
//  A frag: lane = (m%16) + 16*halfA; elems 0..7  = K[halfA*16 + 0..7]
//                                    elems 8..15 = K[halfA*16 + 8..15]
//  B frag: lane = n + 16*(k/16);     elems 0..15 = K[(k/16)*16 + 0..15]
//  C frag: lanes 0-15: N=lane, M=vr; lanes 16-31: N=lane-16, M=8+vr
// All fragment storage below is "fragment-major": one v16bf (32 B) per lane,
// so loads/stores are 2 x b128 with no per-element packing.
// ---------------------------------------------------------------------------

__device__ __forceinline__ v8f wmma_bf16(v16bf a, v16bf b, v8f c) {
  return __builtin_amdgcn_wmma_f32_16x16x32_bf16(
      /*neg_a=*/false, a, /*neg_b=*/false, b,
      /*c_mod=*/(short)0, c, /*reuse_a=*/false, /*reuse_b=*/false);
}

// A (16x32) fragment from a row-major bf16 source (two 16-B vector loads).
__device__ __forceinline__ v16bf load_a_rowmajor(const __bf16* base, int ld, int lane) {
  int r = lane & 15, half = lane >> 4;
  const __bf16* p = base + (size_t)r * ld + half * 16;
  union { v16bf v; v8bf h[2]; } u;
  u.h[0] = *(const v8bf*)(p);
  u.h[1] = *(const v8bf*)(p + 8);
  return u.v;
}

// ---------------- K1: pack weights fp32 -> bf16 B-fragment-major ----------------
// Wpk[z][kt(32)][ng(64)][lane(32)] : lane (c,kh) holds W[kt*32+kh*16+i][ng*16+c]
__global__ __launch_bounds__(256) void pack_w_kernel(
    const float* __restrict__ Wq, const float* __restrict__ Wk,
    const float* __restrict__ Wv, v16bf* __restrict__ Wpk) {
  int flat = blockIdx.x * 256 + threadIdx.x;   // 3*32*64*32 = 196608 total
  int lane = flat & 31;
  int ng   = (flat >> 5) & 63;
  int kt   = (flat >> 11) & 31;
  int z    = flat >> 16;
  const float* W = (z == 0) ? Wq : (z == 1) ? Wk : Wv;
  int c = lane & 15, kh = lane >> 4;
  const float* src = W + (size_t)(kt * 32 + kh * 16) * D_ + ng * 16 + c;
  v16bf v;
#pragma unroll
  for (int i = 0; i < 16; ++i) v[i] = (__bf16)src[(size_t)i * D_];
  Wpk[flat] = v;
}

// ---------------- K2: QKV projection GEMM + bias + ReLU ----------------
// Out[m,n] = relu(sum_k x[b,k,t] * W[k,n] + bias[n]),  m = b*T + t.
// 128x128 block tile, BK=64 (2 wmma k-steps per barrier), 8 waves 4(M)x2(N).
// B fragments come directly from packed global (L2-resident), A tile staged
// into fragment-major LDS. Q,K written [B,T,D]; V written transposed [B,D,T].

#define BM 128
#define BN 128
#define BK 64

__global__ __launch_bounds__(256) void qkv_gemm_kernel(
    const float* __restrict__ x, const v16bf* __restrict__ Wpk,
    const float* __restrict__ bq, const float* __restrict__ bk,
    const float* __restrict__ bv,
    __bf16* __restrict__ Qo, __bf16* __restrict__ Ko, __bf16* __restrict__ Vt) {
  __shared__ v16bf Afrag[2][8][32];   // [kstep][m-frag][lane], 16 KB

  int z = blockIdx.z;
  const v16bf* WpkZ = Wpk + (size_t)z * 32 * 64 * 32;
  const float* bias = (z == 0) ? bq : (z == 1) ? bk : bv;

  int tid = threadIdx.x, lane = tid & 31, w = tid >> 5;
  int m0 = blockIdx.y * BM;            // global row = b*T + t (tile within one b)
  int n0 = blockIdx.x * BN;
  int bi = m0 / T_;
  int t0 = m0 % T_;
  const float* xb = x + (size_t)bi * D_ * T_;   // [D][T]

  int wm = (w & 3) * 32, wn = (w >> 2) * 64;
  v8f acc[2][4] = {};

  for (int k0 = 0; k0 < D_; k0 += BK) {
    // Stage A tile (128 rows x 64 k) into fragment-major LDS.
    // Task (kk8, tt): 8 coalesced float loads along t, one 16-B LDS store.
#pragma unroll
    for (int j = 0; j < 4; ++j) {
      int task = tid + j * 256;        // 1024 tasks
      int tt  = task & 127;
      int kk8 = task >> 7;             // k-octet 0..7
      const float* src = xb + (size_t)(k0 + kk8 * 8) * T_ + t0 + tt;
      v8bf pk;
#pragma unroll
      for (int i = 0; i < 8; ++i) pk[i] = (__bf16)src[(size_t)i * T_];
      if (k0 + BK < D_) __builtin_prefetch(src + (size_t)BK * T_, 0, 0);
      int halfA = (kk8 >> 1) & 1;
      int hh    = kk8 & 1;
      int ks    = kk8 >> 2;
      int dlane = (tt & 15) + 16 * halfA;
      int mf    = tt >> 4;
      *((v8bf*)&Afrag[ks][mf][dlane] + hh) = pk;
    }
    __syncthreads();

#pragma unroll
    for (int ks = 0; ks < 2; ++ks) {
      v16bf a0 = Afrag[ks][(w & 3) * 2 + 0][lane];
      v16bf a1 = Afrag[ks][(w & 3) * 2 + 1][lane];
      int kt = (k0 >> 5) + ks;
#pragma unroll
      for (int ni = 0; ni < 4; ++ni) {
        int ng = ((n0 + wn) >> 4) + ni;
        v16bf bb = WpkZ[(size_t)(kt * 64 + ng) * 32 + lane];
        acc[0][ni] = wmma_bf16(a0, bb, acc[0][ni]);
        acc[1][ni] = wmma_bf16(a1, bb, acc[1][ni]);
      }
    }
    __syncthreads();
  }

  int c = lane & 15, half = lane >> 4;
  if (z < 2) {
    __bf16* out = (z == 0) ? Qo : Ko;        // [B*T][D]
#pragma unroll
    for (int mi = 0; mi < 2; ++mi) {
#pragma unroll
      for (int ni = 0; ni < 4; ++ni) {
        int gn = n0 + wn + ni * 16 + c;
        float bsv = bias[gn];
        int gmb = m0 + wm + mi * 16 + half * 8;
#pragma unroll
        for (int vr = 0; vr < 8; ++vr) {
          float v = fmaxf(acc[mi][ni][vr] + bsv, 0.0f);
          out[(size_t)(gmb + vr) * D_ + gn] = (__bf16)v;
        }
      }
    }
  } else {
    // V transposed: Vt[b][d][t]; octet of vr = 8 consecutive t -> 16-B store
#pragma unroll
    for (int mi = 0; mi < 2; ++mi) {
#pragma unroll
      for (int ni = 0; ni < 4; ++ni) {
        int gn = n0 + wn + ni * 16 + c;      // d
        float bsv = bias[gn];
        int tloc = t0 + wm + mi * 16 + half * 8;
        v8bf pk;
#pragma unroll
        for (int vr = 0; vr < 8; ++vr)
          pk[vr] = (__bf16)fmaxf(acc[mi][ni][vr] + bsv, 0.0f);
        *(v8bf*)(Vt + ((size_t)bi * D_ + gn) * T_ + tloc) = pk;
      }
    }
  }
}

// ---------------- K3: attention, LDS-resident fragment-major K/V ----------------
// One workgroup per (h,b). K fragments (for Q*K^T) and V fragments (for P*V)
// staged as pure 32-B block copies. Online softmax per 32-key chunk.

__global__ __launch_bounds__(256) void attn_kernel(
    const __bf16* __restrict__ Q, const __bf16* __restrict__ K,
    const __bf16* __restrict__ Vt, const float* __restrict__ x,
    float* __restrict__ Y) {
  extern __shared__ char smem_raw[];
  v16bf* Kf = (v16bf*)smem_raw;          // [32 key-group][2 kstep][32 lane] 64 KB
  v16bf* Vf = Kf + 2048;                 // [16 chunk][4 nf][32 lane]        64 KB
  v16bf* Pf = Vf + 2048;                 // [8 wave][32 lane]                 8 KB

  int tid = threadIdx.x, lane = tid & 31, w = tid >> 5;
  int bh = blockIdx.x;
  int h = bh & (H_ - 1);
  int b = bh >> 4;
  int c = lane & 15, half = lane >> 4;

  // Stage K: lane (c,kh) of frag (tg,ks) = K[tg*16+c][ks*32+kh*16 .. +15] (32 B)
  const __bf16* Kg = K + ((size_t)b * T_) * D_ + h * DH_;
#pragma unroll
  for (int j = 0; j < 8; ++j) {
    int task = tid + j * 256;            // 2048 tasks
    int l  = task & 31;
    int ks = (task >> 5) & 1;
    int tg = task >> 6;
    int cc = l & 15, kh = l >> 4;
    Kf[(size_t)(tg * 2 + ks) * 32 + l] =
        *(const v16bf*)(Kg + (size_t)(tg * 16 + cc) * D_ + ks * 32 + kh * 16);
  }
  // Stage V: lane (c,kh) of frag (ch,nf) = Vt[.. nf*16+c ..][ch*32+kh*16 .. +15]
  const __bf16* Vg = Vt + ((size_t)b * D_ + h * DH_) * T_;
#pragma unroll
  for (int j = 0; j < 8; ++j) {
    int task = tid + j * 256;            // 2048 tasks
    int l  = task & 31;
    int nf = (task >> 5) & 3;
    int ch = task >> 7;
    int cc = l & 15, kh = l >> 4;
    Vf[(size_t)(ch * 4 + nf) * 32 + l] =
        *(const v16bf*)(Vg + (size_t)(nf * 16 + cc) * T_ + ch * 32 + kh * 16);
  }
  __syncthreads();

  const __bf16* Qg = Q + ((size_t)b * T_) * D_ + h * DH_;
  __bf16* Pb = (__bf16*)&Pf[(size_t)w * 32];
  int hk = (c >> 3) & 1;                 // P-transpose constants
  int eb = c & 7;

  for (int it = 0; it < 4; ++it) {
    int q0 = (w + 8 * it) * 16;          // Q row tile base
    v16bf qa0 = load_a_rowmajor(Qg + (size_t)q0 * D_ + 0,  D_, lane);
    v16bf qa1 = load_a_rowmajor(Qg + (size_t)q0 * D_ + 32, D_, lane);

    v8f o0 = {}, o1 = {}, o2 = {}, o3 = {};
    float mrow[8], lrow[8];
#pragma unroll
    for (int vr = 0; vr < 8; ++vr) { mrow[vr] = -__builtin_inff(); lrow[vr] = 0.0f; }

    for (int ch = 0; ch < 16; ++ch) {
      // S = Q * K^T for 32 keys: two 16x16 fragments
      v8f s0 = {}, s1 = {};
      s0 = wmma_bf16(qa0, Kf[(size_t)((2 * ch + 0) * 2 + 0) * 32 + lane], s0);
      s0 = wmma_bf16(qa1, Kf[(size_t)((2 * ch + 0) * 2 + 1) * 32 + lane], s0);
      s1 = wmma_bf16(qa0, Kf[(size_t)((2 * ch + 1) * 2 + 0) * 32 + lane], s1);
      s1 = wmma_bf16(qa1, Kf[(size_t)((2 * ch + 1) * 2 + 1) * 32 + lane], s1);

      // online softmax per row (row = half*8 + vr; 16 lanes share a row)
#pragma unroll
      for (int vr = 0; vr < 8; ++vr) {
        float sa = s0[vr] * 0.125f;      // 1/sqrt(64)
        float sb = s1[vr] * 0.125f;
        float mx = fmaxf(sa, sb);
        mx = fmaxf(mx, __shfl_xor(mx, 1, 32));
        mx = fmaxf(mx, __shfl_xor(mx, 2, 32));
        mx = fmaxf(mx, __shfl_xor(mx, 4, 32));
        mx = fmaxf(mx, __shfl_xor(mx, 8, 32));
        float mn = fmaxf(mrow[vr], mx);
        float corr = __expf(mrow[vr] - mn);
        float p0 = __expf(sa - mn);
        float p1 = __expf(sb - mn);
        float rs = p0 + p1;
        rs += __shfl_xor(rs, 1, 32);
        rs += __shfl_xor(rs, 2, 32);
        rs += __shfl_xor(rs, 4, 32);
        rs += __shfl_xor(rs, 8, 32);
        lrow[vr] = lrow[vr] * corr + rs;
        mrow[vr] = mn;
        o0[vr] *= corr; o1[vr] *= corr; o2[vr] *= corr; o3[vr] *= corr;
        // transpose P (C layout) -> A fragment layout in LDS
        int m = half * 8 + vr;
        Pb[(size_t)(m) * 16 + c]      = (__bf16)p0;   // k = c
        Pb[(size_t)(m + 16) * 16 + c] = (__bf16)p1;   // k = 16+c
      }
      // O += P(16x32) * V(32x64)
      v16bf pa = Pf[(size_t)w * 32 + lane];
      o0 = wmma_bf16(pa, Vf[(size_t)(ch * 4 + 0) * 32 + lane], o0);
      o1 = wmma_bf16(pa, Vf[(size_t)(ch * 4 + 1) * 32 + lane], o1);
      o2 = wmma_bf16(pa, Vf[(size_t)(ch * 4 + 2) * 32 + lane], o2);
      o3 = wmma_bf16(pa, Vf[(size_t)(ch * 4 + 3) * 32 + lane], o3);
    }

    // epilogue: 1/l, residual, write pre-BN Y ([B][D][T]); vr-octet = 8
    // consecutive t -> 32-B vector load/store per fragment column.
    const float* xb = x + ((size_t)b * D_ + h * DH_) * T_ + q0 + half * 8;
    float* yb       = Y + ((size_t)b * D_ + h * DH_) * T_ + q0 + half * 8;
    float inv[8];
#pragma unroll
    for (int vr = 0; vr < 8; ++vr) inv[vr] = 1.0f / lrow[vr];
    v8f res;
    v8f xv;
    xv = *(const v8f*)(xb + (size_t)(c)      * T_);
#pragma unroll
    for (int vr = 0; vr < 8; ++vr) res[vr] = o0[vr] * inv[vr] + xv[vr];
    *(v8f*)(yb + (size_t)(c)      * T_) = res;
    xv = *(const v8f*)(xb + (size_t)(16 + c) * T_);
#pragma unroll
    for (int vr = 0; vr < 8; ++vr) res[vr] = o1[vr] * inv[vr] + xv[vr];
    *(v8f*)(yb + (size_t)(16 + c) * T_) = res;
    xv = *(const v8f*)(xb + (size_t)(32 + c) * T_);
#pragma unroll
    for (int vr = 0; vr < 8; ++vr) res[vr] = o2[vr] * inv[vr] + xv[vr];
    *(v8f*)(yb + (size_t)(32 + c) * T_) = res;
    xv = *(const v8f*)(xb + (size_t)(48 + c) * T_);
#pragma unroll
    for (int vr = 0; vr < 8; ++vr) res[vr] = o3[vr] * inv[vr] + xv[vr];
    *(v8f*)(yb + (size_t)(48 + c) * T_) = res;
  }
}

// ---------------- K4: BatchNorm batch stats (one block per channel) ----------------

__global__ __launch_bounds__(256) void bn_stats_kernel(
    const float* __restrict__ Y, float* __restrict__ meanArr,
    float* __restrict__ rstdArr) {
  __shared__ float rs[256], rss[256];
  int d = blockIdx.x, tid = threadIdx.x;
  float s = 0.0f, ss = 0.0f;
  for (int idx = tid; idx < B_ * T_; idx += 256) {
    int b = idx >> 9;
    int t = idx & 511;
    float v = Y[((size_t)b * D_ + d) * T_ + t];
    s += v; ss += v * v;
  }
  rs[tid] = s; rss[tid] = ss;
  __syncthreads();
  for (int st = 128; st > 0; st >>= 1) {
    if (tid < st) { rs[tid] += rs[tid + st]; rss[tid] += rss[tid + st]; }
    __syncthreads();
  }
  if (tid == 0) {
    float inv = 1.0f / (float)(B_ * T_);
    float mean = rs[0] * inv;
    float var = rss[0] * inv - mean * mean;
    meanArr[d] = mean;
    rstdArr[d] = rsqrtf(var + 1e-5f);
  }
}

// ---------------- K5: normalize in place + gamma/beta ----------------

__global__ __launch_bounds__(256) void bn_apply_kernel(
    float* __restrict__ Y, const float* __restrict__ meanArr,
    const float* __restrict__ rstdArr, const float* __restrict__ gamma,
    const float* __restrict__ beta) {
  size_t idx = (size_t)blockIdx.x * 256 + threadIdx.x;
  int d = (int)((idx / T_) % D_);
  float v = Y[idx];
  Y[idx] = (v - meanArr[d]) * rstdArr[d] * gamma[d] + beta[d];
}

// ---------------- host ----------------

extern "C" void kernel_launch(void* const* d_in, const int* in_sizes, int n_in,
                              void* d_out, int out_size, void* d_ws, size_t ws_size,
                              hipStream_t stream) {
  (void)in_sizes; (void)n_in; (void)out_size; (void)ws_size;
  const float* x     = (const float*)d_in[0];
  const float* Wq    = (const float*)d_in[1];
  const float* bq    = (const float*)d_in[2];
  const float* Wk    = (const float*)d_in[3];
  const float* bk    = (const float*)d_in[4];
  const float* Wv    = (const float*)d_in[5];
  const float* bv    = (const float*)d_in[6];
  const float* gamma = (const float*)d_in[7];
  const float* beta  = (const float*)d_in[8];
  float* Y = (float*)d_out;   // pre-BN scratch, finalized in place

  char* ws = (char*)d_ws;
  const size_t wpk_bytes = (size_t)3 * 32 * 64 * 32 * sizeof(v16bf);  // 6 MB
  const size_t act_elems = (size_t)B_ * T_ * D_;
  v16bf* Wpk  = (v16bf*)ws;
  __bf16* Qb  = (__bf16*)(ws + wpk_bytes);
  __bf16* Kb  = Qb + act_elems;
  __bf16* Vtb = Kb + act_elems;                 // transposed [B][D][T]
  float* meanArr = (float*)(ws + wpk_bytes + 3 * act_elems * sizeof(__bf16));
  float* rstdArr = meanArr + D_;

  // K1: weights -> bf16, fragment-major
  pack_w_kernel<<<768, 256, 0, stream>>>(Wq, Wk, Wv, Wpk);

  // K2: QKV projections (z selects Q/K/V)
  dim3 ggrid(D_ / BN, (B_ * T_) / BM, 3);
  qkv_gemm_kernel<<<ggrid, 256, 0, stream>>>(x, Wpk, bq, bk, bv, Qb, Kb, Vtb);

  // K3: attention + residual
  size_t smem = (size_t)(2048 + 2048 + 256) * sizeof(v16bf);  // 139264 B
  attn_kernel<<<H_ * B_, 256, smem, stream>>>(Qb, Kb, Vtb, x, Y);

  // K4/K5: BatchNorm (training-mode batch stats)
  bn_stats_kernel<<<D_, 256, 0, stream>>>(Y, meanArr, rstdArr);
  bn_apply_kernel<<<(int)(((size_t)B_ * D_ * T_) / 256), 256, 0, stream>>>(
      Y, meanArr, rstdArr, gamma, beta);
}
